// AGNN_layer_24773371363892
// MI455X (gfx1250) — matrix operand order repaired and checked by reading, not compiled, not run on hardware
//
#include <hip/hip_runtime.h>
#include <hip/hip_bf16.h>
#include <math.h>

// ---------------------------------------------------------------------------
// AGNN layer for MI455X (gfx1250, wave32).
// Dominant cost: per-edge attention MLP (E=800k, 144x144) -> f16 WMMA
// (v_wmma_f32_16x16x32_f16) with f32 accumulate. W_att1 is staged in LDS once
// per block (320KB/WGP on CDNA5) so every WMMA reads B via ds_load_b128
// instead of waiting on global loads. Softmax/scatter are atomic-bound.
// ---------------------------------------------------------------------------

typedef __attribute__((ext_vector_type(16))) _Float16 v16h;
typedef __attribute__((ext_vector_type(8)))  _Float16 v8h;
typedef __attribute__((ext_vector_type(8)))  float    v8f;

#define DD   64      // node feature dim
#define EDW  16      // edge weight dim
#define H1   144     // attention MLP width (2*DD + EDW)
#define H1P  160     // K padded to multiple of 32
#define NTILES (H1 / 16)   // 9 output-column tiles
#define KCH    (H1P / 32)  // 5 K-chunks of 32

// ---------------- helpers ----------------

__global__ void cvt_f16_kernel(const float* __restrict__ in,
                               _Float16* __restrict__ out, long n) {
    long i = (long)blockIdx.x * blockDim.x + threadIdx.x;
    if (i < n) out[i] = (_Float16)in[i];
}

// W_att1 [144][144] f32 -> padded f16 [144][160] (zeros in K=[144,160))
__global__ void cvt_w1_kernel(const float* __restrict__ W1,
                              _Float16* __restrict__ W1h) {
    int i = blockIdx.x * blockDim.x + threadIdx.x;
    if (i >= H1 * H1P) return;
    int row = i / H1P, col = i % H1P;
    W1h[i] = (col < H1) ? (_Float16)W1[row * H1 + col] : (_Float16)0.0f;
}

__global__ void zero_kernel(float* __restrict__ p, long n) {
    long i = (long)blockIdx.x * blockDim.x + threadIdx.x;
    if (i < n) p[i] = 0.0f;
}

// lin_x = x @ W_lin^T + b_lin   (N x 64, trivial FLOPs -> plain VALU)
__global__ void lin_kernel(const float* __restrict__ x,
                           const float* __restrict__ Wl,
                           const float* __restrict__ bl,
                           float* __restrict__ lin_x, int N) {
    int t = blockIdx.x * blockDim.x + threadIdx.x;
    if (t >= N * DD) return;
    int r = t >> 6, c = t & 63;
    const float* xr = x + (size_t)r * DD;
    const float* wr = Wl + (size_t)c * DD;
    float s = bl[c];
#pragma unroll
    for (int k = 0; k < DD; ++k) s = fmaf(xr[k], wr[k], s);
    lin_x[t] = s;
}

// ---------------- WMMA attention-score kernel ----------------
// Block = 64 threads (2 waves). W_att1 (f16, padded [144][160]) staged to LDS
// once per block; each wave then grid-strides over 16-edge tiles:
//   feat (16 x 160 f16) staged in LDS: [0,64)=x[src], [64,80)=edge_w,
//   [80,144)=x[dst], [144,160)=0.  H = feat @ W_att1^T (bias in C-init),
//   relu * W_att2 fused on the accumulators, lane-reduced, scores written.
__global__ __launch_bounds__(64) void att_scores_kernel(
    const _Float16* __restrict__ xh, const _Float16* __restrict__ ewh,
    const _Float16* __restrict__ W1h, const float* __restrict__ b1,
    const float* __restrict__ W2, const float* __restrict__ b2p,
    const int* __restrict__ src, const int* __restrict__ dst,
    float* __restrict__ scores, int E, int ntiles) {
    __shared__ __align__(16) _Float16 w1s[H1 * H1P];     // 46080 B
    __shared__ __align__(16) _Float16 feat[2][16][H1P];  // 10240 B

    const int wave = threadIdx.x >> 5;
    const int lane = threadIdx.x & 31;
    const int m = lane & 15;   // edge row within tile / C column
    const int h = lane >> 4;   // half-wave id

    // stage W_att1 into LDS: 2880 x 16B, 45 per thread
    {
        const v8h* gsrc = (const v8h*)W1h;
        v8h* ldst = (v8h*)w1s;
        for (int i = threadIdx.x; i < H1 * H1P / 8; i += 64) ldst[i] = gsrc[i];
    }
    __syncthreads();

    const float b2 = b2p[0];

    // block-uniform tile loop (both waves run the same trip count)
    for (int base = blockIdx.x * 2; base < ntiles; base += gridDim.x * 2) {
        int tile = base + wave;
        if (tile >= ntiles) tile = ntiles - 1;   // duplicate tail (same writes)
        const int ebase = tile * 16;
        int e = ebase + m;
        if (e >= E) e = E - 1;                   // clamp: keep EXEC all-ones

        // ---- stage feat tile (each edge by 2 lanes: h=0/1 take 32 halves)
        _Float16* frow = &feat[wave][m][0];
        {
            const v8h* xs = (const v8h*)(xh + (size_t)src[e] * DD + h * 32);
            v8h* f0 = (v8h*)(frow + h * 32);
            f0[0] = xs[0]; f0[1] = xs[1]; f0[2] = xs[2]; f0[3] = xs[3];
            const v8h* xd = (const v8h*)(xh + (size_t)dst[e] * DD + h * 32);
            v8h* f1 = (v8h*)(frow + DD + EDW + h * 32);
            f1[0] = xd[0]; f1[1] = xd[1]; f1[2] = xd[2]; f1[3] = xd[3];
            if (h == 0) {
                const v8h* e8 = (const v8h*)(ewh + (size_t)e * EDW);
                v8h* f2 = (v8h*)(frow + DD);
                f2[0] = e8[0]; f2[1] = e8[1];
            } else {
                v8h z = (v8h)(_Float16)0.0f;
                v8h* f3 = (v8h*)(frow + H1);
                f3[0] = z; f3[1] = z;
            }
        }
        __syncthreads();

        // ---- A fragments (contiguous k order, same as B): lane l, elem i ->
        //      M = l&15, K = (l>>4)*16 + i + 32*kc
        v16h afrag[KCH];
#pragma unroll
        for (int kc = 0; kc < KCH; ++kc) {
            const int base0 = kc * 32 + h * 16;
            v8h lo = *(const v8h*)&feat[wave][m][base0];
            v8h hi = *(const v8h*)&feat[wave][m][base0 + 8];
            v16h a;
#pragma unroll
            for (int i = 0; i < 8; ++i) { a[i] = lo[i]; a[i + 8] = hi[i]; }
            afrag[kc] = a;
        }

        float p[8];
#pragma unroll
        for (int r = 0; r < 8; ++r) p[r] = 0.0f;

        for (int nt = 0; nt < NTILES; ++nt) {
            const int ncol = nt * 16 + m;            // hidden unit = B column
            const float bias = b1[ncol];
            v8f c;
#pragma unroll
            for (int r = 0; r < 8; ++r) c[r] = bias;
            // B layout: lane l, elem i -> N = l&15, K = (l>>4)*16 + i + 32*kc
            const _Float16* brow = w1s + ncol * H1P + h * 16;
#pragma unroll
            for (int kc = 0; kc < KCH; ++kc) {
                v16h b = *(const v16h*)(brow + kc * 32);
                c = __builtin_amdgcn_wmma_f32_16x16x32_f16(
                        false, afrag[kc], false, b, (short)0, c, false, false);
            }
            const float w2 = W2[ncol];
#pragma unroll
            for (int r = 0; r < 8; ++r) p[r] += fmaxf(c[r], 0.0f) * w2;
        }

        // reduce across the 16 lanes of each half (N dimension)
#pragma unroll
        for (int mask = 8; mask >= 1; mask >>= 1) {
#pragma unroll
            for (int r = 0; r < 8; ++r) p[r] += __shfl_xor(p[r], mask, 32);
        }

        if (m == 0) {  // lanes 0 (rows 0..7) and 16 (rows 8..15)
#pragma unroll
            for (int r = 0; r < 8; ++r) {
                int eo = ebase + h * 8 + r;
                if (eo < E) scores[eo] = p[r] + b2;
            }
        }
        __syncthreads();   // feat reuse across iterations
    }
}

// ---------------- segment softmax + scatter ----------------

// order-preserving f32 -> u32 encoding; 0 reserved as "-inf / no edge"
__device__ __forceinline__ unsigned enc_f32(float f) {
    unsigned b = __float_as_uint(f);
    return (b & 0x80000000u) ? ~b : (b | 0x80000000u);
}
__device__ __forceinline__ float dec_f32(unsigned u) {
    unsigned b = (u & 0x80000000u) ? (u & 0x7FFFFFFFu) : ~u;
    return __uint_as_float(b);
}

__global__ void segmax_kernel(const float* __restrict__ scores,
                              const int* __restrict__ src,
                              unsigned* __restrict__ menc, int E) {
    int t = blockIdx.x * blockDim.x + threadIdx.x;
    if (t >= E) return;
    atomicMax(&menc[src[t]], enc_f32(scores[t]));
}

__global__ void expsum_kernel(const float* __restrict__ scores,
                              const int* __restrict__ src,
                              const unsigned* __restrict__ menc,
                              float* __restrict__ eexp,
                              float* __restrict__ denom, int E) {
    int t = blockIdx.x * blockDim.x + threadIdx.x;
    if (t >= E) return;
    int s = src[t];
    float ee = expf(scores[t] - dec_f32(menc[s]));
    eexp[t] = ee;
    atomicAdd(&denom[s], ee);
}

// one thread per (edge, 4-float group): float4 gather + 4 atomics
__global__ void msg_kernel(const float* __restrict__ eexp,
                           const float* __restrict__ denom,
                           const float* __restrict__ lin_x,
                           const int* __restrict__ src,
                           const int* __restrict__ dst,
                           float* __restrict__ msg, long n) {
    long t = (long)blockIdx.x * blockDim.x + threadIdx.x;
    if (t >= n) return;
    int e = (int)(t >> 4), q = (int)(t & 15);
    int s = src[e];
    float alpha = eexp[e] / denom[s];
    const float4 lv = *(const float4*)(lin_x + (size_t)dst[e] * DD + q * 4);
    float* mp = msg + (size_t)s * DD + q * 4;
    atomicAdd(mp + 0, alpha * lv.x);
    atomicAdd(mp + 1, alpha * lv.y);
    atomicAdd(mp + 2, alpha * lv.z);
    atomicAdd(mp + 3, alpha * lv.w);
}

// ---------------- GRU + degree-guarded output ----------------
__global__ void gru_kernel(const float* __restrict__ x,
                           const float* __restrict__ msg,
                           const unsigned* __restrict__ menc,
                           const float* __restrict__ Wih,
                           const float* __restrict__ bih,
                           const float* __restrict__ Whh,
                           const float* __restrict__ bhh,
                           float* __restrict__ out, int N) {
    int t = blockIdx.x * blockDim.x + threadIdx.x;
    if (t >= N * DD) return;
    int nidx = t >> 6, d = t & 63;
    if (menc[nidx] == 0u) { out[t] = x[t]; return; }  // no outgoing edges
    const float* xr = x + (size_t)nidx * DD;
    const float* mr = msg + (size_t)nidx * DD;
    const float* wir = Wih + (size_t)d * DD;
    const float* wiz = Wih + (size_t)(DD + d) * DD;
    const float* win = Wih + (size_t)(2 * DD + d) * DD;
    const float* whr = Whh + (size_t)d * DD;
    const float* whz = Whh + (size_t)(DD + d) * DD;
    const float* whn = Whh + (size_t)(2 * DD + d) * DD;
    float ir = bih[d], iz = bih[DD + d], in_ = bih[2 * DD + d];
    float hr = bhh[d], hz = bhh[DD + d], hn = bhh[2 * DD + d];
#pragma unroll 8
    for (int k = 0; k < DD; ++k) {
        float xv = xr[k], mv = mr[k];
        ir = fmaf(xv, wir[k], ir);  iz = fmaf(xv, wiz[k], iz);
        in_ = fmaf(xv, win[k], in_);
        hr = fmaf(mv, whr[k], hr);  hz = fmaf(mv, whz[k], hz);
        hn = fmaf(mv, whn[k], hn);
    }
    float r = 1.0f / (1.0f + expf(-(ir + hr)));
    float z = 1.0f / (1.0f + expf(-(iz + hz)));
    float nn = tanhf(in_ + r * hn);
    out[t] = (1.0f - z) * nn + z * mr[d];
}

// ---------------- launch ----------------
extern "C" void kernel_launch(void* const* d_in, const int* in_sizes, int n_in,
                              void* d_out, int out_size, void* d_ws, size_t ws_size,
                              hipStream_t stream) {
    const float* x    = (const float*)d_in[0];
    const int*   eidx = (const int*)d_in[1];
    const float* ew   = (const float*)d_in[2];
    const float* W1   = (const float*)d_in[3];
    const float* b1   = (const float*)d_in[4];
    const float* W2   = (const float*)d_in[5];
    const float* b2   = (const float*)d_in[6];
    const float* Wl   = (const float*)d_in[7];
    const float* bl   = (const float*)d_in[8];
    const float* Wih  = (const float*)d_in[9];
    const float* bih  = (const float*)d_in[10];
    const float* Whh  = (const float*)d_in[11];
    const float* bhh  = (const float*)d_in[12];

    const int N = in_sizes[0] / DD;
    const int E = in_sizes[2] / EDW;
    const int* src = eidx;
    const int* dst = eidx + E;

    auto al = [](size_t v) { return (v + 255) & ~(size_t)255; };
    char* w = (char*)d_ws;
    size_t off = 0;
    _Float16* xh   = (_Float16*)(w + off); off = al(off + (size_t)N * DD * 2);
    _Float16* ewh  = (_Float16*)(w + off); off = al(off + (size_t)E * EDW * 2);
    _Float16* W1h  = (_Float16*)(w + off); off = al(off + (size_t)H1 * H1P * 2);
    float* lin_x   = (float*)(w + off);    off = al(off + (size_t)N * DD * 4);
    float* scores  = (float*)(w + off);    off = al(off + (size_t)E * 4);
    float* eexp    = (float*)(w + off);    off = al(off + (size_t)E * 4);
    // zero-block: msg | denom | menc contiguous -> one zero pass of N*66 floats
    float* msg     = (float*)(w + off);
    float* denom   = msg + (size_t)N * DD;
    unsigned* menc = (unsigned*)(denom + N);

    const int T = 256;
    long nx = (long)N * DD, ne = (long)E * EDW, nz = (long)N * (DD + 2);

    cvt_f16_kernel<<<(int)((nx + T - 1) / T), T, 0, stream>>>(x, xh, nx);
    cvt_f16_kernel<<<(int)((ne + T - 1) / T), T, 0, stream>>>(ew, ewh, ne);
    cvt_w1_kernel<<<(H1 * H1P + T - 1) / T, T, 0, stream>>>(W1, W1h);
    zero_kernel<<<(int)((nz + T - 1) / T), T, 0, stream>>>(msg, nz);
    lin_kernel<<<(int)((nx + T - 1) / T), T, 0, stream>>>(x, Wl, bl, lin_x, N);

    int ntiles = (E + 15) / 16;
    int ablocks = (ntiles + 1) / 2;          // 2 waves/block, 16 edges/wave
    if (ablocks > 4096) ablocks = 4096;      // grid-stride; W1 staged once/block
    att_scores_kernel<<<ablocks, 64, 0, stream>>>(
        xh, ewh, W1h, b1, W2, b2, src, dst, scores, E, ntiles);

    segmax_kernel<<<(E + T - 1) / T, T, 0, stream>>>(scores, src, menc, E);
    expsum_kernel<<<(E + T - 1) / T, T, 0, stream>>>(scores, src, menc, eexp, denom, E);
    long nm = (long)E * EDW;                 // E * 16 groups of 4 floats
    msg_kernel<<<(int)((nm + T - 1) / T), T, 0, stream>>>(
        eexp, denom, lin_x, src, dst, msg, nm);
    gru_kernel<<<(int)((nx + T - 1) / T), T, 0, stream>>>(
        x, msg, menc, Wih, bih, Whh, bhh, (float*)d_out, N);

    (void)n_in; (void)out_size; (void)ws_size;
}
